// SelfAttention_8546984919496
// MI455X (gfx1250) — compile-verified, matrix operand order checked by reading, not run on hardware
//
#include <hip/hip_runtime.h>

// Self-attention fused kernel for gfx1250 (MI455X).
// out = softmax(Q K^T) V  with  Q=XWq, K=XWk, V=XWv, heads = groups of 8 rows.
// Refactor: out_tile = (A_blockdiag * X_tile) * Wv  -> V never materialized.
// This revision: 2x2 register-blocked phase-6 GEMM (1 fragment load per WMMA)
// and asynchronous Wv panel staging via GLOBAL_LOAD_ASYNC_TO_LDS_B128.

constexpr int kDim     = 1024;   // model dim
constexpr int kKeyD    = 64;     // key dim
constexpr int kRows    = 128;    // rows per workgroup (16 groups of 8)
constexpr int kThreads = 512;    // 16 wave32s
constexpr int kXS      = kDim  + 8;  // padded LDS row stride (bank-conflict-free frags)
constexpr int kQS      = kKeyD + 8;

// LDS arena layout (bytes)
constexpr int kXBytes   = kRows * kXS * 2;          // 264192: X strip (later A*X)
constexpr int kQBytes   = kRows * kQS * 2;          // 18432
constexpr int kOffQ     = 0;                        // region2-relative
constexpr int kOffK     = kQBytes;                  // 18432
constexpr int kOffS     = 2 * kQBytes;              // 36864 (scores, f32)
constexpr int kOffW     = kOffS + kRows * 16 * 4;   // 45056 (softmax w, f32)
constexpr int kR2Bytes  = kOffW + kRows * 8 * 4;    // 49152
// Phase-6 B panels (128 cols x 64 k, double buffered) overlap dead Q/K/S region
constexpr int kBStride  = 72;                       // elements per panel row (64 + 8 pad)
constexpr int kBPanelB  = 128 * kBStride * 2;       // 18432 bytes per buffer
constexpr int kSmem     = kXBytes + kR2Bytes;       // 313344 total (<= 320K WGP LDS)

typedef __attribute__((ext_vector_type(16))) __bf16 v16bf;
typedef __attribute__((ext_vector_type(8)))  float  v8f;

static __device__ __forceinline__ unsigned short f2bf(float f) {
  union { float f; unsigned int u; } c; c.f = f;
  return (unsigned short)((c.u + 0x7FFFu + ((c.u >> 16) & 1u)) >> 16);  // RNE
}
static __device__ __forceinline__ float bf2f(unsigned short h) {
  union { unsigned int u; float f; } c; c.u = (unsigned int)h << 16;
  return c.f;
}

// Load one 16x32 bf16 fragment in the CDNA5 16-bit A-matrix layout from a
// row-major ushort buffer (works for LDS and global).
//   lanes 0-15  : row = row0+lane,    K = {k0..k0+7,  k0+16..k0+23}
//   lanes 16-31 : row = row0+lane-16, K = {k0+8..k0+15, k0+24..k0+31}
template <typename T>
static __device__ __forceinline__ v16bf load_frag(const T* buf, int row0, int k0,
                                                  int stride, int lane) {
  int e = (row0 + (lane & 15)) * stride + k0 + ((lane >> 4) << 3);
  union { uint4 q[2]; v16bf v; } u;
  u.q[0] = *(const uint4*)(buf + e);
  u.q[1] = *(const uint4*)(buf + e + 16);
  return u.v;
}

// CDNA5 async memory->LDS copy, 16 bytes per lane (ISA 15.18.3 op 98).
// VDST holds the LDS byte address; ASYNCcnt tracks completion.
static __device__ __forceinline__ void async_ld_b128(void* lds_ptr, const void* gptr) {
  unsigned int lds = (unsigned int)(uintptr_t)lds_ptr;   // low 32 bits = LDS address
  unsigned long long ga = (unsigned long long)(uintptr_t)gptr;
  asm volatile("global_load_async_to_lds_b128 %0, %1, off"
               :: "v"(lds), "v"(ga) : "memory");
}
static __device__ __forceinline__ void wait_async0() {
  asm volatile("s_wait_asynccnt 0" ::: "memory");
}

// ---------------- prep: W^T in bf16 (B operand is loaded column-major) -----
__global__ void transpose_to_bf16(const float* __restrict__ src,
                                  unsigned short* __restrict__ dst,
                                  int nrows, int ncols) {
  // dst[n * nrows + k] = bf16(src[k * ncols + n])
  int idx = blockIdx.x * blockDim.x + threadIdx.x;
  if (idx >= nrows * ncols) return;
  int n = idx / nrows;
  int k = idx - n * nrows;
  dst[idx] = f2bf(src[(size_t)k * ncols + n]);
}

// ---------------- fused attention ------------------------------------------
__global__ __launch_bounds__(kThreads) void attn_fused(
    const float* __restrict__ X,
    const unsigned short* __restrict__ WqT,   // [kKeyD][kDim] bf16
    const unsigned short* __restrict__ WkT,   // [kKeyD][kDim] bf16
    const unsigned short* __restrict__ WvT,   // [kDim][kDim]  bf16
    float* __restrict__ Out) {
  __shared__ __align__(16) char smem[kSmem];
  unsigned short* Xs = (unsigned short*)smem;
  char* region2      = smem + kXBytes;
  unsigned short* Qs = (unsigned short*)(region2 + kOffQ);
  unsigned short* Ks = (unsigned short*)(region2 + kOffK);
  float*          Sb = (float*)(region2 + kOffS);
  float*          Wb = (float*)(region2 + kOffW);

  const int tid  = threadIdx.x;
  const int lane = tid & 31;
  const int wave = tid >> 5;
  const size_t row0 = (size_t)blockIdx.x * kRows;

  // ---- Phase 1: stage X strip -> LDS bf16 (single HBM read of X) ----------
  {
    const float4* Xg = (const float4*)(X + row0 * kDim);
    #pragma unroll 4
    for (int i = 0; i < (kRows * kDim / 4) / kThreads; ++i) {
      int f4 = i * kThreads + tid;
      float4 v = Xg[f4];
      int r  = f4 >> 8;          // 256 float4 per row
      int c4 = f4 & 255;
      uint2 p;
      p.x = (unsigned int)f2bf(v.x) | ((unsigned int)f2bf(v.y) << 16);
      p.y = (unsigned int)f2bf(v.z) | ((unsigned int)f2bf(v.w) << 16);
      *(uint2*)&Xs[r * kXS + c4 * 4] = p;
    }
  }
  __syncthreads();

  // ---- Phase 2: Q = X Wq, K = X Wk (WMMA, f32 accum -> bf16 in LDS) -------
  {
    const unsigned short* BT = (wave < 8) ? WqT : WkT;
    unsigned short* QK       = (wave < 8) ? Qs  : Ks;
    const int mt = wave & 7;                     // M-tile of this wave
    for (int nt = 0; nt < 4; ++nt) {             // 4 N-tiles cover kKeyD
      v8f acc = {};
      #pragma unroll 4
      for (int ks = 0; ks < kDim / 32; ++ks) {
        v16bf a = load_frag(Xs, mt * 16, ks * 32, kXS, lane);
        v16bf b = load_frag(BT, nt * 16, ks * 32, kDim, lane);
        acc = __builtin_amdgcn_wmma_f32_16x16x32_bf16(false, a, false, b,
                                                      (short)0, acc, false, false);
      }
      const int m0 = mt * 16 + ((lane >> 4) << 3);
      const int n  = nt * 16 + (lane & 15);
      #pragma unroll
      for (int v = 0; v < 8; ++v) QK[(m0 + v) * kQS + n] = f2bf(acc[v]);
    }
  }
  __syncthreads();

  // ---- Phase 3: scores S = Q K^T per 16-row block (2 groups/WMMA) ---------
  if (wave < 8) {                                // wave-uniform branch: EXEC full
    const int b = wave;
    v8f acc = {};
    #pragma unroll
    for (int ks = 0; ks < kKeyD / 32; ++ks) {
      v16bf a  = load_frag(Qs, b * 16, ks * 32, kQS, lane);
      v16bf bt = load_frag(Ks, b * 16, ks * 32, kQS, lane);  // B = K^T col-major == K rows
      acc = __builtin_amdgcn_wmma_f32_16x16x32_bf16(false, a, false, bt,
                                                    (short)0, acc, false, false);
    }
    const int m0 = b * 16 + ((lane >> 4) << 3);
    const int n  = lane & 15;
    #pragma unroll
    for (int v = 0; v < 8; ++v) Sb[(m0 + v) * 16 + n] = acc[v];
  }
  __syncthreads();

  // ---- Phase 4: softmax over each row's own group (8 wide, unscaled) ------
  if (tid < kRows) {
    const int r    = tid;
    const int base = ((r & 15) < 8) ? 0 : 8;     // diagonal 8x8 block of the 16x16
    float s[8], mx = -3.4e38f;
    #pragma unroll
    for (int j = 0; j < 8; ++j) { s[j] = Sb[r * 16 + base + j]; mx = fmaxf(mx, s[j]); }
    float sum = 0.f;
    #pragma unroll
    for (int j = 0; j < 8; ++j) { s[j] = __expf(s[j] - mx); sum += s[j]; }
    const float inv = 1.0f / sum;
    #pragma unroll
    for (int j = 0; j < 8; ++j) Wb[r * 8 + j] = s[j] * inv;
  }
  __syncthreads();

  // ---- Phase 5: in-place row mix  X <- A_blockdiag * X  -------------------
  for (int i = 0; i < 32; ++i) {
    const int g = i >> 1;                        // same group across all lanes
    const int c = ((i & 1) << 9) | tid;
    float x[8];
    #pragma unroll
    for (int m = 0; m < 8; ++m) x[m] = bf2f(Xs[(g * 8 + m) * kXS + c]);
    #pragma unroll
    for (int r = 0; r < 8; ++r) {
      float o = 0.f;
      #pragma unroll
      for (int m = 0; m < 8; ++m) o += Wb[(g * 8 + r) * 8 + m] * x[m];
      Xs[(g * 8 + r) * kXS + c] = f2bf(o);
    }
  }
  __syncthreads();

  // ---- Phase 6: Out = (A X) Wv  (dominant GEMM, 128x1024 @ 1024x1024) -----
  // 8 panels of 128 cols; K split into 16 chunks of 64. Each chunk of Wv is
  // async-copied memory->LDS (double buffered) and shared by all 16 waves, so
  // each Wv element is fetched from L2 once per WG. Waves are 2x2 register
  // blocked: 4 fragment loads feed 4 WMMAs.
  {
    const int m0 = (wave & 3) * 32;              // 2 M-tiles: m0, m0+16
    const int n0 = (wave >> 2) * 32;             // 2 N-tiles within 128-col panel

    unsigned short* Bbuf0 = (unsigned short*)region2;
    unsigned short* Bbuf1 = (unsigned short*)(region2 + kBPanelB);

    // async stage chunk 0 into buffer 0
    #pragma unroll
    for (int i = 0; i < 2; ++i) {
      int u  = i * kThreads + tid;
      int c  = u >> 3;                           // 8 x b128 per 64-elem row
      int ko = (u & 7) << 3;
      async_ld_b128(Bbuf0 + c * kBStride + ko, WvT + (size_t)c * kDim + ko);
    }
    wait_async0();
    __syncthreads();

    v8f a00 = {}, a01 = {}, a10 = {}, a11 = {};
    for (int q = 0; q < 128; ++q) {              // 8 panels x 16 k-chunks of 64
      const int p  = q >> 4;
      const int kc = q & 15;
      unsigned short* Bc = (q & 1) ? Bbuf1 : Bbuf0;
      unsigned short* Bn = (q & 1) ? Bbuf0 : Bbuf1;
      if (q + 1 < 128) {                         // async stage next chunk
        const int pn = (q + 1) >> 4;
        const int kn = (q + 1) & 15;
        #pragma unroll
        for (int i = 0; i < 2; ++i) {
          int u  = i * kThreads + tid;
          int c  = u >> 3;
          int ko = (u & 7) << 3;
          async_ld_b128(Bn + c * kBStride + ko,
                        WvT + (size_t)(pn * 128 + c) * kDim + kn * 64 + ko);
        }
      }
      #pragma unroll
      for (int ks = 0; ks < 2; ++ks) {           // 64-wide chunk = 2 WMMA k-steps
        v16bf A0 = load_frag(Xs, m0,      kc * 64 + ks * 32, kXS, lane);
        v16bf A1 = load_frag(Xs, m0 + 16, kc * 64 + ks * 32, kXS, lane);
        v16bf B0 = load_frag(Bc, n0,      ks * 32, kBStride, lane);
        v16bf B1 = load_frag(Bc, n0 + 16, ks * 32, kBStride, lane);
        a00 = __builtin_amdgcn_wmma_f32_16x16x32_bf16(false, A0, false, B0, (short)0, a00, false, false);
        a01 = __builtin_amdgcn_wmma_f32_16x16x32_bf16(false, A0, false, B1, (short)0, a01, false, false);
        a10 = __builtin_amdgcn_wmma_f32_16x16x32_bf16(false, A1, false, B0, (short)0, a10, false, false);
        a11 = __builtin_amdgcn_wmma_f32_16x16x32_bf16(false, A1, false, B1, (short)0, a11, false, false);
      }
      if (kc == 15) {                            // panel done: write 2x2 16x16 tiles
        const size_t or0 = row0 + m0 + ((lane >> 4) << 3);
        const int    oc0 = p * 128 + n0 + (lane & 15);
        #pragma unroll
        for (int v = 0; v < 8; ++v) {
          Out[(or0 + v)      * kDim + oc0]      = a00[v];
          Out[(or0 + v)      * kDim + oc0 + 16] = a01[v];
          Out[(or0 + 16 + v) * kDim + oc0]      = a10[v];
          Out[(or0 + 16 + v) * kDim + oc0 + 16] = a11[v];
        }
        a00 = (v8f){}; a01 = (v8f){}; a10 = (v8f){}; a11 = (v8f){};
      }
      wait_async0();                             // my async stores landed in LDS
      __syncthreads();                           // everyone's staged data visible
    }
  }
}

// ---------------- host side ------------------------------------------------
extern "C" void kernel_launch(void* const* d_in, const int* in_sizes, int n_in,
                              void* d_out, int out_size, void* d_ws, size_t ws_size,
                              hipStream_t stream) {
  const float* X  = (const float*)d_in[0];   // [N, 1024]
  const float* Wq = (const float*)d_in[1];   // [1024, 64]
  const float* Wk = (const float*)d_in[2];   // [1024, 64]
  const float* Wv = (const float*)d_in[3];   // [1024, 1024]
  float* Out = (float*)d_out;

  unsigned short* WqT = (unsigned short*)d_ws;            // [64][1024] bf16
  unsigned short* WkT = WqT + kKeyD * kDim;               // [64][1024] bf16
  unsigned short* WvT = WkT + kKeyD * kDim;               // [1024][1024] bf16

  const int nRows = in_sizes[0] / kDim;                   // N = 131072

  transpose_to_bf16<<<(kKeyD * kDim + 255) / 256, 256, 0, stream>>>(Wq, WqT, kDim, kKeyD);
  transpose_to_bf16<<<(kKeyD * kDim + 255) / 256, 256, 0, stream>>>(Wk, WkT, kDim, kKeyD);
  transpose_to_bf16<<<(kDim  * kDim + 255) / 256, 256, 0, stream>>>(Wv, WvT, kDim, kDim);

  attn_fused<<<nRows / kRows, kThreads, 0, stream>>>(X, WqT, WkT, WvT, Out);
}
